// Memory_67972152427212
// MI455X (gfx1250) — matrix operand order repaired and checked by reading, not compiled, not run on hardware
//
#include <hip/hip_runtime.h>
#include <hip/hip_bf16.h>
#include <math.h>

#define B_ 64
#define U_ 32
#define N_ 512
#define D_ 256
#define K_ (D_ + N_)   // 768 = concatenated contraction dim [Win ; W*scale]
#define KSTEPS (K_ / 32)

typedef __attribute__((ext_vector_type(16))) __bf16 v16bf;
typedef __attribute__((ext_vector_type(8)))  __bf16 v8bf;
typedef __attribute__((ext_vector_type(8)))  float  v8f;

union Frag16 { v16bf v; v8bf h[2]; };

// ---------------------------------------------------------------------------
// Kernel 1: spectral radius via power iteration, one workgroup per u.
// scale[u] = sr[u] / rho(W_u). W_u is L2-resident (32 MB total).
// ---------------------------------------------------------------------------
__global__ __launch_bounds__(512) void esn_power_iter(const float* __restrict__ W,
                                                      const float* __restrict__ sr,
                                                      float* __restrict__ scale) {
  __shared__ float xs[N_];
  __shared__ float ys[N_];
  __shared__ float red[16];
  const int u    = blockIdx.x;
  const int tid  = threadIdx.x;     // 0..511 == N_
  const int lane = tid & 31;
  const int wave = tid >> 5;        // 0..15
  const float* Wu = W + (size_t)u * N_ * N_;

  xs[tid] = 1.0f;                   // deterministic start vector
  __syncthreads();

  const int ITERS = 32, TAIL = 12;
  float logsum = 0.0f;
  for (int it = 0; it < ITERS; ++it) {
    // y = W x : each wave owns 32 rows; lanes stride the columns (coalesced)
    for (int r = 0; r < 32; ++r) {
      const int row = wave * 32 + r;
      const float* wrow = Wu + (size_t)row * N_;
      float acc = 0.0f;
      #pragma unroll
      for (int j = lane; j < N_; j += 32) acc += wrow[j] * xs[j];
      #pragma unroll
      for (int off = 16; off > 0; off >>= 1) acc += __shfl_xor(acc, off, 32);
      if (lane == 0) ys[row] = acc;
    }
    __syncthreads();
    // ||y||
    float v  = ys[tid];
    float sq = v * v;
    #pragma unroll
    for (int off = 16; off > 0; off >>= 1) sq += __shfl_xor(sq, off, 32);
    if (lane == 0) red[wave] = sq;
    __syncthreads();
    float n2 = 0.0f;
    #pragma unroll
    for (int w = 0; w < 16; ++w) n2 += red[w];
    const float g = sqrtf(n2);
    xs[tid] = v * (1.0f / g);
    if (it >= ITERS - TAIL) logsum += logf(g);   // geometric-mean growth
    __syncthreads();
  }
  if (tid == 0) scale[u] = sr[u] / expf(logsum / (float)TAIL);
}

// ---------------------------------------------------------------------------
// Kernel 2: build K-major (transposed) bf16 hi/lo planes of B = [Win ; W*scale]
// Bt[u][n][k], pitch K_. 32x32 LDS tile transpose, coalesced both sides.
// ---------------------------------------------------------------------------
__global__ __launch_bounds__(256) void esn_pack_b(const float* __restrict__ W,
                                                  const float* __restrict__ Win,
                                                  const float* __restrict__ scale,
                                                  __bf16* __restrict__ Bth,
                                                  __bf16* __restrict__ Btl) {
  __shared__ float tile[32][33];
  const int u  = blockIdx.z;
  const int k0 = blockIdx.y * 32;
  const int n0 = blockIdx.x * 32;
  const int tn = threadIdx.x & 31;   // col within tile (source, coalesced)
  const int tk = threadIdx.x >> 5;   // 0..7
  const float s = scale[u];
  #pragma unroll
  for (int i = 0; i < 4; ++i) {
    const int k = k0 + tk + 8 * i;   // tiles never straddle k==D_ (both 32-aligned)
    float v = (k < D_) ? Win[((size_t)u * D_ + k) * N_ + (n0 + tn)]
                       : W  [((size_t)u * N_ + (k - D_)) * N_ + (n0 + tn)] * s;
    tile[tk + 8 * i][tn] = v;
  }
  __syncthreads();
  const int tk2 = threadIdx.x & 31;  // k within tile (dest, coalesced)
  const int tn2 = threadIdx.x >> 5;
  #pragma unroll
  for (int i = 0; i < 4; ++i) {
    const int n = tn2 + 8 * i;
    const float  v = tile[tk2][n];
    const __bf16 h = (__bf16)v;
    const __bf16 l = (__bf16)(v - (float)h);
    const size_t d = ((size_t)u * N_ + (n0 + n)) * K_ + (k0 + tk2);
    Bth[d] = h;
    Btl[d] = l;
  }
}

// ---------------------------------------------------------------------------
// Kernel 3: fused WMMA GEMM + bias + tanh + leaky update.
// Block tile = 64(M) x 64(N); 8 waves as 4(M) x 2(N32); each wave owns TWO
// independent 16x16 f32 accumulators (shared A fragment -> 2 WMMA chains).
// Split-bf16: c += Ah*Bh + Ah*Bl + Al*Bh  (~fp32 accuracy at bf16 WMMA rate).
// B fragments are double-buffered (software pipeline) to hide load latency.
// ---------------------------------------------------------------------------
__global__ __launch_bounds__(256) void esn_wmma(const float* __restrict__ X,
                                                const float* __restrict__ state,
                                                const __bf16* __restrict__ Bth,
                                                const __bf16* __restrict__ Btl,
                                                const float* __restrict__ bias,
                                                const float* __restrict__ lr,
                                                float* __restrict__ out) {
  const int u    = blockIdx.y;
  const int n0   = blockIdx.x * 64;
  const int tid  = threadIdx.x;
  const int lane = tid & 31;
  const int wave = tid >> 5;             // 0..7
  const int mw   = (wave & 3) * 16;      // wave M offset (rows = batch b)
  const int nw   = n0 + (wave >> 2) * 32;// wave N offset (owns nw..nw+31)
  const int half = lane >> 4;            // 0: lanes 0-15, 1: lanes 16-31
  const int l15  = lane & 15;

  // B rows (K-major) for the two 16-column groups this wave owns
  const __bf16* brh0 = Bth + ((size_t)u * N_ + (nw + l15))      * K_ + half * 16;
  const __bf16* brl0 = Btl + ((size_t)u * N_ + (nw + l15))      * K_ + half * 16;
  const __bf16* brh1 = Bth + ((size_t)u * N_ + (nw + 16 + l15)) * K_ + half * 16;
  const __bf16* brl1 = Btl + ((size_t)u * N_ + (nw + 16 + l15)) * K_ + half * 16;
  // A row for this lane: logical A[u][m][k] = k<D ? X[m][u][k] : state[m][u][k-D]
  const float* xrow = X     + ((size_t)(mw + l15) * U_ + u) * D_;
  const float* srow = state + ((size_t)(mw + l15) * U_ + u) * N_;

  v8f c0 = {}, c1 = {};
  Frag16 bh[2][2], bl[2][2];             // [buffer][column-group]

  // prologue: fill buffer 0 with K-step 0
  bh[0][0].h[0] = *(const v8bf*)(brh0);     bh[0][0].h[1] = *(const v8bf*)(brh0 + 8);
  bl[0][0].h[0] = *(const v8bf*)(brl0);     bl[0][0].h[1] = *(const v8bf*)(brl0 + 8);
  bh[0][1].h[0] = *(const v8bf*)(brh1);     bh[0][1].h[1] = *(const v8bf*)(brh1 + 8);
  bl[0][1].h[0] = *(const v8bf*)(brl1);     bl[0][1].h[1] = *(const v8bf*)(brl1 + 8);

  #pragma unroll 2
  for (int ks = 0; ks < KSTEPS; ++ks) {
    const int cur = ks & 1, nxt = cur ^ 1;
    const int k0  = ks * 32;
    // issue next K-step's B loads (clamped on last step: redundant, never OOB)
    const int kn = (ks + 1 < KSTEPS) ? (k0 + 32) : k0;
    bh[nxt][0].h[0] = *(const v8bf*)(brh0 + kn); bh[nxt][0].h[1] = *(const v8bf*)(brh0 + kn + 8);
    bl[nxt][0].h[0] = *(const v8bf*)(brl0 + kn); bl[nxt][0].h[1] = *(const v8bf*)(brl0 + kn + 8);
    bh[nxt][1].h[0] = *(const v8bf*)(brh1 + kn); bh[nxt][1].h[1] = *(const v8bf*)(brh1 + kn + 8);
    bl[nxt][1].h[0] = *(const v8bf*)(brl1 + kn); bl[nxt][1].h[1] = *(const v8bf*)(brl1 + kn + 8);
    __builtin_prefetch(brh0 + k0 + 256, 0, 1);   // global_prefetch_b8

    // ---- A fragment (on-the-fly f32 -> bf16 hi/lo split), shared by c0/c1 ----
    const float* asrc = (k0 < D_) ? (xrow + k0 + half * 16)
                                  : (srow + (k0 - D_) + half * 16);
    const v8f a0 = *(const v8f*)(asrc);
    const v8f a1 = *(const v8f*)(asrc + 8);
    Frag16 a_h, a_l;
    #pragma unroll
    for (int i = 0; i < 8; ++i) {
      const __bf16 h0 = (__bf16)a0[i];
      const __bf16 h1 = (__bf16)a1[i];
      a_h.v[i]     = h0;
      a_h.v[8 + i] = h1;
      a_l.v[i]     = (__bf16)(a0[i] - (float)h0);
      a_l.v[8 + i] = (__bf16)(a1[i] - (float)h1);
    }

    // ---- two independent accumulator chains (XDL ILP) ----
    c0 = __builtin_amdgcn_wmma_f32_16x16x32_bf16(false, a_h.v, false, bh[cur][0].v,
                                                 (short)0, c0, false, false);
    c1 = __builtin_amdgcn_wmma_f32_16x16x32_bf16(false, a_h.v, false, bh[cur][1].v,
                                                 (short)0, c1, false, false);
    c0 = __builtin_amdgcn_wmma_f32_16x16x32_bf16(false, a_h.v, false, bl[cur][0].v,
                                                 (short)0, c0, false, false);
    c1 = __builtin_amdgcn_wmma_f32_16x16x32_bf16(false, a_h.v, false, bl[cur][1].v,
                                                 (short)0, c1, false, false);
    c0 = __builtin_amdgcn_wmma_f32_16x16x32_bf16(false, a_l.v, false, bh[cur][0].v,
                                                 (short)0, c0, false, false);
    c1 = __builtin_amdgcn_wmma_f32_16x16x32_bf16(false, a_l.v, false, bh[cur][1].v,
                                                 (short)0, c1, false, false);
  }

  // ---- epilogue: acc VGPR r -> (M = r + half*8, N = l15) ----
  const float lru = lr[u];
  const int   na  = nw + l15;
  const int   nb  = nw + 16 + l15;
  const float bia = bias[(size_t)u * N_ + na];
  const float bib = bias[(size_t)u * N_ + nb];
  #pragma unroll
  for (int r = 0; r < 8; ++r) {
    const int    b  = mw + r + half * 8;           // batch row
    const size_t oa = ((size_t)b * U_ + u) * N_ + na;
    const size_t ob = ((size_t)b * U_ + u) * N_ + nb;
    const float sa = state[oa];
    const float sb = state[ob];
    out[oa] = (1.0f - lru) * sa + lru * tanhf(c0[r] + bia);
    out[ob] = (1.0f - lru) * sb + lru * tanhf(c1[r] + bib);
  }
}

// ---------------------------------------------------------------------------
extern "C" void kernel_launch(void* const* d_in, const int* in_sizes, int n_in,
                              void* d_out, int out_size, void* d_ws, size_t ws_size,
                              hipStream_t stream) {
  (void)in_sizes; (void)n_in; (void)out_size; (void)ws_size;
  const float* X     = (const float*)d_in[0];
  const float* state = (const float*)d_in[1];
  const float* sr    = (const float*)d_in[2];
  const float* lr    = (const float*)d_in[3];
  const float* W     = (const float*)d_in[4];
  const float* Win   = (const float*)d_in[5];
  const float* bias  = (const float*)d_in[6];
  float* out = (float*)d_out;

  char*  ws    = (char*)d_ws;
  float* scale = (float*)ws;                          // U_ floats
  size_t off   = 256;
  __bf16* Bth = (__bf16*)(ws + off); off += (size_t)U_ * N_ * K_ * 2;  // 24 MB
  __bf16* Btl = (__bf16*)(ws + off);                                   // 24 MB

  esn_power_iter<<<U_, 512, 0, stream>>>(W, sr, scale);

  dim3 gb(N_ / 32, K_ / 32, U_);
  esn_pack_b<<<gb, 256, 0, stream>>>(W, Win, scale, Bth, Btl);

  dim3 gw(N_ / 64, U_);
  esn_wmma<<<gw, 256, 0, stream>>>(X, state, Bth, Btl, bias, lr, out);
}
